// LSTM_35150012351217
// MI455X (gfx1250) — matrix operand order repaired and checked by reading, not compiled, not run on hardware
//
#include <hip/hip_runtime.h>
#include <hip/hip_bf16.h>

#define H     1024
#define SEQ   128
#define BATCH 256
#define NCLS  10

typedef __attribute__((ext_vector_type(16))) _Float16 v16h;
typedef __attribute__((ext_vector_type(8)))  _Float16 v8h;
typedef __attribute__((ext_vector_type(8)))  float    v8f;

__device__ __forceinline__ float sigmoid_f(float x) {
    return 1.0f / (1.0f + __expf(-x));
}
__device__ __forceinline__ float tanh_f(float x) {
    // tanh via sigmoid; saturates correctly at +/-1 for large |x|
    return 2.0f / (1.0f + __expf(-2.0f * x)) - 1.0f;
}

// Zero-init h ping buffer (as dwords) and the grid-barrier counter.
__global__ void init_state(unsigned int* __restrict__ h0, int nh,
                           unsigned int* __restrict__ ctr) {
    int i = blockIdx.x * blockDim.x + threadIdx.x;
    if (i < nh) h0[i] = 0u;
    if (i == 0) *ctr = 0u;
}

// Convert the four recurrent weight matrices (H x H, f32) to f16, gate-major.
__global__ void convert_weights(const float* __restrict__ wg, const float* __restrict__ wi,
                                const float* __restrict__ wf, const float* __restrict__ wo,
                                _Float16* __restrict__ W16) {
    int idx  = blockIdx.x * blockDim.x + threadIdx.x;          // 0 .. H*H-1
    int gate = blockIdx.y;                                     // 0..3
    const float* src = (gate == 0) ? wg : (gate == 1) ? wi : (gate == 2) ? wf : wo;
    W16[(size_t)gate * H * H + idx] = (_Float16)src[idx];
}

// Device-scope grid barrier (monotonic counter, one arrival per block per step).
__device__ __forceinline__ void grid_barrier(unsigned int* ctr, unsigned int target) {
    __threadfence();                 // release: make h_out stores visible (global_wb)
    __syncthreads();
    if (threadIdx.x == 0) {
        __hip_atomic_fetch_add(ctr, 1u, __ATOMIC_RELEASE, __HIP_MEMORY_SCOPE_AGENT);
        while (__hip_atomic_load(ctr, __ATOMIC_ACQUIRE, __HIP_MEMORY_SCOPE_AGENT) < target) {
            __builtin_amdgcn_s_sleep(8);
        }
    }
    __syncthreads();
    __threadfence();                 // acquire: drop stale WGP$/L0 lines (global_inv)
}

// Persistent cooperative LSTM: the full 128-step scan in ONE kernel.
// Grid = (BATCH/32, H/64) = 128 blocks of 128 threads (4 waves).
// Wave w computes gate w for a 32(batch) x 64(hidden) tile: 2 M-tiles x 4 N-tiles,
// B fragments reused across both M-tiles. c lives in registers for all 128 steps.
__global__ __launch_bounds__(128, 1)
void lstm_persistent(const _Float16* __restrict__ W16,   // [4][H][H] f16
                     _Float16* __restrict__ hb0,         // [B][H] f16 (holds h_0 = 0)
                     _Float16* __restrict__ hb1,         // [B][H] f16
                     const float* __restrict__ x,        // [B][SEQ]
                     const float* __restrict__ wx_g, const float* __restrict__ bias_g,
                     const float* __restrict__ wx_i, const float* __restrict__ bias_i,
                     const float* __restrict__ wx_f, const float* __restrict__ bias_f,
                     const float* __restrict__ wx_o, const float* __restrict__ bias_o,
                     unsigned int* __restrict__ barrier_ctr) {
    __shared__ float acc_lds[4][32][64];

    const int tid  = threadIdx.x;
    const int wave = tid >> 5;      // gate index 0..3 (g,i,f,o)
    const int lane = tid & 31;
    const int lh   = lane & 15;
    const int lg   = lane >> 4;     // 0 or 1
    const int b0   = blockIdx.x * 32;
    const int j0   = blockIdx.y * 64;
    const unsigned int nblocks = gridDim.x * gridDim.y;

    const _Float16* Wg = W16 + (size_t)wave * (H * H);

    // Per-thread cell state: e = tid + i*128 -> (m = e>>6, n = e&63), fixed for all t.
    float creg[16];
    #pragma unroll
    for (int i = 0; i < 16; ++i) creg[i] = 0.0f;

    const _Float16* hin  = hb0;
    _Float16*       hout = hb1;

    for (int t = 0; t < SEQ; ++t) {
        v8f acc[2][4] = {};
        const _Float16* aRow0 = hin + (size_t)(b0 + lh) * H + lg * 8;
        const _Float16* aRow1 = hin + (size_t)(b0 + 16 + lh) * H + lg * 8;

        #pragma unroll 2
        for (int k0 = 0; k0 < H; k0 += 32) {
            // Two A fragments (16x32 f16 each): contiguous 16B loads per lane.
            union { v16h v; v8h h2[2]; } A0, A1;
            A0.h2[0] = *(const v8h*)(aRow0 + k0);
            A0.h2[1] = *(const v8h*)(aRow0 + k0 + 16);
            A1.h2[0] = *(const v8h*)(aRow1 + k0);
            A1.h2[1] = *(const v8h*)(aRow1 + k0 + 16);

            #pragma unroll
            for (int nt = 0; nt < 4; ++nt) {
                // B fragment (32x16 f16): B[:,n] = W row (j0+nt*16+n); two 16B loads at k0+8*lg and k0+16+8*lg.
                const _Float16* bRow = Wg + (size_t)(j0 + nt * 16 + lh) * H + k0 + lg * 8;
                const v16h B = __builtin_shufflevector(*(const v8h*)(bRow), *(const v8h*)(bRow + 16), 0, 1, 2, 3, 4, 5, 6, 7, 8, 9, 10, 11, 12, 13, 14, 15);
                acc[0][nt] = __builtin_amdgcn_wmma_f32_16x16x32_f16(
                    false, A0.v, false, B, (short)0, acc[0][nt], false, false);
                acc[1][nt] = __builtin_amdgcn_wmma_f32_16x16x32_f16(
                    false, A1.v, false, B, (short)0, acc[1][nt], false, false);
            }
        }

        // Spill per-gate accumulators to LDS per the 16x16 f32 C/D layout.
        #pragma unroll
        for (int mt = 0; mt < 2; ++mt) {
            #pragma unroll
            for (int nt = 0; nt < 4; ++nt) {
                #pragma unroll
                for (int v = 0; v < 8; ++v) {
                    acc_lds[wave][mt * 16 + v + lg * 8][nt * 16 + lh] = acc[mt][nt][v];
                }
            }
        }
        __syncthreads();

        // Pointwise cell update: 2048 elements, 16 per thread, c in registers.
        #pragma unroll
        for (int i = 0; i < 16; ++i) {
            const int e = tid + i * 128;
            const int m = e >> 6, n = e & 63;
            const int b = b0 + m, j = j0 + n;
            const float xb = x[b * SEQ + t];     // INPUT_DIM == 1: rank-1 x-term
            const float ag = acc_lds[0][m][n] + xb * wx_g[j] + bias_g[j];
            const float ai = acc_lds[1][m][n] + xb * wx_i[j] + bias_i[j];
            const float af = acc_lds[2][m][n] + xb * wx_f[j] + bias_f[j];
            const float ao = acc_lds[3][m][n] + xb * wx_o[j] + bias_o[j];
            const float cnew = tanh_f(ag) * sigmoid_f(ai) + creg[i] * sigmoid_f(af);
            creg[i] = cnew;
            hout[(size_t)b * H + j] = (_Float16)(tanh_f(cnew) * sigmoid_f(ao));
        }

        // Ping-pong h and synchronize the whole grid before the next step.
        { const _Float16* tmp = hin; hin = hout; hout = (_Float16*)tmp; }
        grid_barrier(barrier_ctr, (unsigned int)(t + 1) * nblocks);
    }
}

// p = h @ w_ph.T + bias_p, then row softmax. One block per batch row.
__global__ __launch_bounds__(256)
void proj_softmax(const _Float16* __restrict__ h, const float* __restrict__ w_ph,
                  const float* __restrict__ bias_p, float* __restrict__ out) {
    __shared__ float red[NCLS][256];
    const int b = blockIdx.x, tid = threadIdx.x;

    float accp[NCLS];
    #pragma unroll
    for (int q = 0; q < NCLS; ++q) accp[q] = 0.0f;

    for (int k = tid; k < H; k += 256) {
        const float hv = (float)h[(size_t)b * H + k];
        #pragma unroll
        for (int q = 0; q < NCLS; ++q) accp[q] += hv * w_ph[q * H + k];
    }
    #pragma unroll
    for (int q = 0; q < NCLS; ++q) red[q][tid] = accp[q];
    __syncthreads();

    for (int s = 128; s > 0; s >>= 1) {
        if (tid < s) {
            #pragma unroll
            for (int q = 0; q < NCLS; ++q) red[q][tid] += red[q][tid + s];
        }
        __syncthreads();
    }

    if (tid == 0) {
        float p[NCLS], mx = -1e30f;
        #pragma unroll
        for (int q = 0; q < NCLS; ++q) {
            p[q] = red[q][0] + bias_p[q];
            mx = fmaxf(mx, p[q]);
        }
        float sum = 0.0f;
        #pragma unroll
        for (int q = 0; q < NCLS; ++q) { p[q] = __expf(p[q] - mx); sum += p[q]; }
        const float inv = 1.0f / sum;
        #pragma unroll
        for (int q = 0; q < NCLS; ++q) out[b * NCLS + q] = p[q] * inv;
    }
}

extern "C" void kernel_launch(void* const* d_in, const int* in_sizes, int n_in,
                              void* d_out, int out_size, void* d_ws, size_t ws_size,
                              hipStream_t stream) {
    const float* x      = (const float*)d_in[0];
    const float* w_gx   = (const float*)d_in[1];
    const float* w_gh   = (const float*)d_in[2];
    const float* bias_g = (const float*)d_in[3];
    const float* w_ix   = (const float*)d_in[4];
    const float* w_ih   = (const float*)d_in[5];
    const float* bias_i = (const float*)d_in[6];
    const float* w_fx   = (const float*)d_in[7];
    const float* w_fh   = (const float*)d_in[8];
    const float* bias_f = (const float*)d_in[9];
    const float* w_ox   = (const float*)d_in[10];
    const float* w_oh   = (const float*)d_in[11];
    const float* bias_o = (const float*)d_in[12];
    const float* w_ph   = (const float*)d_in[13];
    const float* bias_p = (const float*)d_in[14];

    // Workspace layout:
    //   [0, 8 MiB)           : W16, 4 gates x H x H f16
    //   [8 MiB, +512 KiB)    : h ping buffer (f16)  -- initial h_0 = 0
    //   [8.5 MiB, +512 KiB)  : h pong buffer (f16)
    //   [9 MiB, +4 B)        : grid-barrier counter
    char* ws = (char*)d_ws;
    _Float16*     W16 = (_Float16*)ws;
    _Float16*     hb0 = (_Float16*)(ws + 8ull * 1024 * 1024);
    _Float16*     hb1 = (_Float16*)(ws + 8ull * 1024 * 1024 + 512ull * 1024);
    unsigned int* ctr = (unsigned int*)(ws + 9ull * 1024 * 1024);

    // Per-call init (ws is poisoned; kernel_launch must be self-contained).
    const int nh = BATCH * H / 2;   // h buffer in dwords
    init_state<<<(nh + 255) / 256, 256, 0, stream>>>((unsigned int*)hb0, nh, ctr);
    convert_weights<<<dim3(H * H / 256, 4), 256, 0, stream>>>(w_gh, w_ih, w_fh, w_oh, W16);

    // Entire 128-step recurrence in one persistent cooperative kernel.
    lstm_persistent<<<dim3(BATCH / 32, H / 64), 128, 0, stream>>>(
        W16, hb0, hb1, x,
        w_gx, bias_g, w_ix, bias_i, w_fx, bias_f, w_ox, bias_o, ctr);

    // After an even number of steps the final h lives in hb0.
    proj_softmax<<<BATCH, 256, 0, stream>>>(hb0, w_ph, bias_p, (float*)d_out);
}
